// MediaPipeLSTM_46402826666511
// MI455X (gfx1250) — matrix operand order repaired and checked by reading, not compiled, not run on hardware
//
#include <hip/hip_runtime.h>
#include <hip/hip_bf16.h>

// ---------------------------------------------------------------------------
// MediaPipe-style 3-stream LSTM scan for MI455X (gfx1250), wave32 + WMMA bf16.
//
// - x-projection fused into step kernel (weights are L2 resident: ~11 MB fp32
//   << 192 MB L2; x streamed once, 527 MB ~= 23 us @ 23.3 TB/s).
// - 2 kernels per timestep (serial chain is the binding constraint).
// - bf16 WMMA (v_wmma_f32_16x16x32_bf16) with f32 accumulation; fp32->bf16
//   conversion during global->LDS staging (v_cvt_pk_bf16_f32 + ds_store_b32).
// - Double-buffered LDS staging via SINGLE base pointer + integer offset so
//   address-space inference keeps ds_load/ds_store.
// - Loop body order: WMMA(chunk c) first, then stage(chunk c+1) — 16-bit WMMA
//   co-executes (TRANS-class) with the staging loads/converts that follow.
// - Staging loads batched into register arrays so global loads cluster into
//   clauses instead of per-element load->wait->convert chains.
// - LDS tiles laid out so each WMMA fragment is two ds_load_b128 per lane
//   per the CDNA5 16-bit A/B VGPR layouts (ISA 7.12.2).
// ---------------------------------------------------------------------------

typedef __bf16 bf16_t;
typedef __attribute__((ext_vector_type(16))) __bf16       v16bf;
typedef __attribute__((ext_vector_type(8)))  float        v8f;
typedef __attribute__((ext_vector_type(4)))  unsigned int u32x4;
typedef __attribute__((ext_vector_type(2)))  float        f32x2;

union FragBF { v16bf v; u32x4 q[2]; };
union PackBF { bf16_t h[2]; unsigned int u; };

#define SEQ_LEN 155
#define IN_DIM  1662
#define HID     256
#define OUTN    50
#define BATCH   512
#define FC1N    (OUTN + 30)

#define AS_LD 40    // bf16 elems per A-tile row (64 rows)  -> 80 B pitch
#define BS_LD 40    // bf16 elems per B-tile row (256 rows)
#define GB_LD 260   // floats per gate-exchange row (32 rows)

#define A_ELEMS     (64 * AS_LD)
#define STAGE_ELEMS (64 * AS_LD + 256 * BS_LD)   // 12800 bf16 = 25600 B
// LDS: 2 staging buffers (51200 B) overlapped with gate-exchange buffer
// (32 rows * 260 floats = 33280 B). Max = 51200 B.
#define SMEM_BYTES (2 * STAGE_ELEMS * 2)

__device__ __forceinline__ float sigmoidf_(float x) { return 1.0f / (1.0f + __expf(-x)); }

// --- branch-free staging of one 32-wide K chunk (fp32 -> packed bf16) -------
// Loads are batched into register arrays first so they issue as clauses and
// the loadcnt waits amortize (instead of per-element load->wait->cvt chains).
__device__ __forceinline__ void stage_full(
    bf16_t* __restrict__ As, bf16_t* __restrict__ Bs,
    const float* __restrict__ A, int lda,
    const float* __restrict__ B, int ldb,
    int row0, int hb0, int k0, int tid)
{
  // A tile: 64 rows x 32 cols = 1024 float2, 4 per thread (one batch)
  f32x2 va[4];
#pragma unroll
  for (int i = 0; i < 4; ++i) {
    int idx = tid + 256 * i;
    int r = idx >> 4, c2 = (idx & 15) << 1;
    va[i] = *(const f32x2*)(A + (size_t)(row0 + r) * lda + k0 + c2);
  }
#pragma unroll
  for (int i = 0; i < 4; ++i) {
    int idx = tid + 256 * i;
    int r = idx >> 4, c2 = (idx & 15) << 1;
    PackBF p; p.h[0] = (bf16_t)va[i].x; p.h[1] = (bf16_t)va[i].y;
    *(unsigned int*)(As + r * AS_LD + c2) = p.u;
  }
  // B tile: 256 gate rows x 32 cols = 4096 float2, 16 per thread
  // (two batches of 8 to bound transient register pressure)
#pragma unroll
  for (int h = 0; h < 2; ++h) {
    f32x2 vb[8];
#pragma unroll
    for (int j = 0; j < 8; ++j) {
      int idx = tid + 256 * (h * 8 + j);
      int lr = idx >> 4, c2 = (idx & 15) << 1;
      int g = ((lr >> 6) << 8) + hb0 + (lr & 63);  // gate-group*256 + hidden
      vb[j] = *(const f32x2*)(B + (size_t)g * ldb + k0 + c2);
    }
#pragma unroll
    for (int j = 0; j < 8; ++j) {
      int idx = tid + 256 * (h * 8 + j);
      int lr = idx >> 4, c2 = (idx & 15) << 1;
      PackBF p; p.h[0] = (bf16_t)vb[j].x; p.h[1] = (bf16_t)vb[j].y;
      *(unsigned int*)(Bs + lr * BS_LD + c2) = p.u;
    }
  }
}

// --- masked tail chunk (only x segment: valid = 1662 % 32 = 30) -------------
__device__ __forceinline__ void stage_tail(
    bf16_t* __restrict__ As, bf16_t* __restrict__ Bs,
    const float* __restrict__ A, int lda,
    const float* __restrict__ B, int ldb,
    int row0, int hb0, int k0, int valid, int tid)
{
#pragma unroll
  for (int i = 0; i < 8; ++i) {
    int idx = tid + 256 * i;
    int r = idx >> 5, c = idx & 31;
    float v = (c < valid) ? A[(size_t)(row0 + r) * lda + k0 + c] : 0.0f;
    As[r * AS_LD + c] = (bf16_t)v;
  }
#pragma unroll
  for (int i = 0; i < 32; ++i) {
    int idx = tid + 256 * i;
    int lr = idx >> 5, c = idx & 31;
    int g = ((lr >> 6) << 8) + hb0 + (lr & 63);
    float v = (c < valid) ? B[(size_t)g * ldb + k0 + c] : 0.0f;
    Bs[lr * BS_LD + c] = (bf16_t)v;
  }
}

// --- one K=32 WMMA step: fragments per CDNA5 16-bit layouts ------------------
__device__ __forceinline__ void mma_step(
    v8f (&acc)[2][4], const bf16_t* __restrict__ As, const bf16_t* __restrict__ Bs,
    int wm, int wn, int lh, int l15)
{
  FragBF a[2], b[4];
#pragma unroll
  for (int mf = 0; mf < 2; ++mf) {
    const bf16_t* p = As + (wm * 32 + mf * 16 + l15) * AS_LD + 8 * lh;
    a[mf].q[0] = *(const u32x4*)p;         // K = 8*lh + 0..7
    a[mf].q[1] = *(const u32x4*)(p + 16);  // K = 16 + 8*lh + 0..7
  }
#pragma unroll
  for (int nf = 0; nf < 4; ++nf) {
    const bf16_t* p = Bs + (wn * 64 + nf * 16 + l15) * BS_LD + 8 * lh;
    b[nf].q[0] = *(const u32x4*)p;         // K = 8*lh + 0..7
    b[nf].q[1] = *(const u32x4*)(p + 16);  // K = 16 + 8*lh + 0..7
  }
#pragma unroll
  for (int mf = 0; mf < 2; ++mf)
#pragma unroll
    for (int nf = 0; nf < 4; ++nf)
      acc[mf][nf] = __builtin_amdgcn_wmma_f32_16x16x32_bf16(
          false, a[mf].v, false, b[nf].v, (short)0, acc[mf][nf], false, false);
}

// ---------------------------------------------------------------------------
// One LSTM-cell tile: gates[64 x 256] = [A0|A1][64 x (K0+K1)] * [B0|B1]^T,
// then pointwise LSTM update for a 64(batch) x 64(hidden) tile.
// Block: 256 threads = 8 waves, 2(M) x 4(N); wave wn owns gate group wn.
// ---------------------------------------------------------------------------
__device__ __forceinline__ void lstm_cell_tile(
    const float* __restrict__ A0, int lda0, const float* __restrict__ B0, int ldb0, int K0,
    const float* __restrict__ A1, int lda1, const float* __restrict__ B1, int ldb1, int K1,
    const float* __restrict__ bih, const float* __restrict__ bhh,
    const float* __restrict__ c_in,
    float* __restrict__ h_out, float* __restrict__ c_out,
    char* smem)
{
  bf16_t* const sb = (bf16_t*)smem;  // single LDS base; buffers = sb + off
  float*  const Gb = (float*)smem;   // overlaps staging; protected by barriers

  const int tid  = threadIdx.x;
  const int lane = tid & 31;
  const int wid  = tid >> 5;
  const int wm   = wid & 1;          // wave row (0..1)  -> rows 32*wm..+31
  const int wn   = wid >> 1;         // wave col (0..3)  -> gate group wn
  const int lh   = lane >> 4;
  const int l15  = lane & 15;
  const int row0 = blockIdx.y * 64;  // batch-row base
  const int hb0  = blockIdx.x * 64;  // hidden-col base

  const int n0full = K0 >> 5;            // full chunks of segment 0
  const int tail   = K0 & 31;            // masked columns in tail chunk
  const int n0     = n0full + (tail ? 1 : 0);
  const int nch    = n0 + (K1 >> 5);     // total chunk count

  // per-chunk dispatcher (branch is per chunk, not per element)
  auto stage_chunk = [&](int c, bf16_t* as, bf16_t* bs) {
    if (c < n0full) {
      stage_full(as, bs, A0, lda0, B0, ldb0, row0, hb0, c << 5, tid);
    } else if (c < n0) {
      stage_tail(as, bs, A0, lda0, B0, ldb0, row0, hb0, n0full << 5, tail, tid);
    } else {
      stage_full(as, bs, A1, lda1, B1, ldb1, row0, hb0, (c - n0) << 5, tid);
    }
  };

  v8f acc[2][4] = {};

  stage_chunk(0, sb, sb + A_ELEMS);  // prologue into buffer 0
  for (int c = 0; c < nch; ++c) {
    __syncthreads();  // chunk c staged & visible; reused buffer free
    // WMMA first: 16-bit WMMAs are TRANS-class and co-execute with the
    // staging loads/converts issued right after.
    const int cb = (c & 1) * STAGE_ELEMS;
    mma_step(acc, sb + cb, sb + cb + A_ELEMS, wm, wn, lh, l15);
    if (c + 1 < nch) {
      const int nb = ((c + 1) & 1) * STAGE_ELEMS;
      stage_chunk(c + 1, sb + nb, sb + nb + A_ELEMS);
    }
  }

  // --- epilogue: exchange gates via LDS in two 32-row phases, then LSTM ---
#pragma unroll
  for (int ph = 0; ph < 2; ++ph) {
    __syncthreads();
    if (wm == ph) {
#pragma unroll
      for (int mf = 0; mf < 2; ++mf)
#pragma unroll
        for (int nf = 0; nf < 4; ++nf)
#pragma unroll
          for (int r = 0; r < 8; ++r)
            Gb[(mf * 16 + r + 8 * lh) * GB_LD + wn * 64 + nf * 16 + l15] =
                acc[mf][nf][r];
    }
    __syncthreads();
#pragma unroll
    for (int i = 0; i < 8; ++i) {
      int idx = tid + 256 * i;
      int r = idx >> 6, n = idx & 63;        // r in 0..31, n in 0..63
      int h  = hb0 + n;
      int gr = row0 + ph * 32 + r;
      float gi = Gb[r * GB_LD +       n] + bih[      h] + bhh[      h];
      float gf = Gb[r * GB_LD +  64 + n] + bih[256 + h] + bhh[256 + h];
      float gg = Gb[r * GB_LD + 128 + n] + bih[512 + h] + bhh[512 + h];
      float go = Gb[r * GB_LD + 192 + n] + bih[768 + h] + bhh[768 + h];
      gi = sigmoidf_(gi); gf = sigmoidf_(gf); go = sigmoidf_(go); gg = tanhf(gg);
      float c = gf * c_in[gr * HID + h] + gi * gg;
      c_out[gr * HID + h] = c;
      h_out[gr * HID + h] = go * tanhf(c);
    }
  }
}

// ---------------------------------------------------------------------------
// Step kernel 1: cell 1, gates = [x_t | h1_prev] x [Wih1 | Whh1]^T + b1
// Grid: (4 hidden tiles, 8 batch tiles), 256 threads.
// ---------------------------------------------------------------------------
__global__ __launch_bounds__(256) void lstm_step1(
    const float* __restrict__ xt,
    const float* __restrict__ Wih1, const float* __restrict__ Whh1,
    const float* __restrict__ bih1, const float* __restrict__ bhh1,
    const float* __restrict__ h1_in, const float* __restrict__ c1_in,
    float* __restrict__ h1_out, float* __restrict__ c1_out)
{
  __shared__ alignas(16) char smem[SMEM_BYTES];
  lstm_cell_tile(xt, IN_DIM, Wih1, IN_DIM, IN_DIM,
                 h1_in, HID, Whh1, HID, HID,
                 bih1, bhh1, c1_in, h1_out, c1_out, smem);
}

// ---------------------------------------------------------------------------
// Step kernel 2: cells 2 and 3 (blockIdx.z selects stream).
// gates_s = [h1_new | h_s] x [Wih2 | Whh2]^T + b2   (shared in2 folded in)
// Grid: (4, 8, 2), 256 threads.
// ---------------------------------------------------------------------------
__global__ __launch_bounds__(256) void lstm_step23(
    const float* __restrict__ h1,
    const float* __restrict__ Wih2, const float* __restrict__ Whh2,
    const float* __restrict__ bih2, const float* __restrict__ bhh2,
    const float* __restrict__ h2_in, const float* __restrict__ c2_in,
    float* __restrict__ h2_out, float* __restrict__ c2_out,
    const float* __restrict__ h3_in, const float* __restrict__ c3_in,
    float* __restrict__ h3_out, float* __restrict__ c3_out)
{
  __shared__ alignas(16) char smem[SMEM_BYTES];
  const float* hs = blockIdx.z ? h3_in : h2_in;
  const float* cs = blockIdx.z ? c3_in : c2_in;
  float* ho = blockIdx.z ? h3_out : h2_out;
  float* co = blockIdx.z ? c3_out : c2_out;
  lstm_cell_tile(h1, HID, Wih2, HID, HID,
                 hs, HID, Whh2, HID, HID,
                 bih2, bhh2, cs, ho, co, smem);
}

// ---------------------------------------------------------------------------
// Tiny MLP head (negligible cost): scalar kernels.
// ---------------------------------------------------------------------------
__global__ void fc1_kernel(const float* __restrict__ h3,
                           const float* __restrict__ W, const float* __restrict__ b,
                           float* __restrict__ out)
{
  int idx = blockIdx.x * blockDim.x + threadIdx.x;
  if (idx >= BATCH * FC1N) return;
  int bi = idx / FC1N, j = idx % FC1N;
  float s = b[j];
#pragma unroll 8
  for (int k = 0; k < HID; ++k) s += h3[bi * HID + k] * W[j * HID + k];
  out[idx] = fmaxf(s, 0.0f);
}

__global__ void fc2_kernel(const float* __restrict__ hid,
                           const float* __restrict__ W, const float* __restrict__ b,
                           float* __restrict__ out)
{
  int idx = blockIdx.x * blockDim.x + threadIdx.x;
  if (idx >= BATCH * OUTN) return;
  int bi = idx / OUTN, j = idx % OUTN;
  float s = b[j];
#pragma unroll 8
  for (int k = 0; k < FC1N; ++k) s += hid[bi * FC1N + k] * W[j * FC1N + k];
  out[idx] = fmaxf(s, 0.0f);
}

// ---------------------------------------------------------------------------
// Launcher: ping-pong state buffers in workspace; 2 kernels per timestep.
// ---------------------------------------------------------------------------
extern "C" void kernel_launch(void* const* d_in, const int* in_sizes, int n_in,
                              void* d_out, int out_size, void* d_ws, size_t ws_size,
                              hipStream_t stream) {
  (void)in_sizes; (void)n_in; (void)out_size; (void)ws_size;

  const float* x    = (const float*)d_in[0];
  const float* Wih1 = (const float*)d_in[1];
  const float* Whh1 = (const float*)d_in[2];
  const float* bih1 = (const float*)d_in[3];
  const float* bhh1 = (const float*)d_in[4];
  const float* Wih2 = (const float*)d_in[5];
  const float* Whh2 = (const float*)d_in[6];
  const float* bih2 = (const float*)d_in[7];
  const float* bhh2 = (const float*)d_in[8];
  const float* Wfc1 = (const float*)d_in[9];
  const float* bfc1 = (const float*)d_in[10];
  const float* Wfc2 = (const float*)d_in[11];
  const float* bfc2 = (const float*)d_in[12];

  float* ws = (float*)d_ws;
  const size_t S = (size_t)BATCH * HID;  // 131072 floats per state buffer
  float* h1b[2] = { ws + 0 * S, ws + 1 * S };
  float* c1b[2] = { ws + 2 * S, ws + 3 * S };
  float* h2b[2] = { ws + 4 * S, ws + 5 * S };
  float* c2b[2] = { ws + 6 * S, ws + 7 * S };
  float* h3b[2] = { ws + 8 * S, ws + 9 * S };
  float* c3b[2] = { ws + 10 * S, ws + 11 * S };
  float* hid = ws + 12 * S;

  // initial states (d2d copies are graph-capture safe)
  hipMemcpyAsync(h1b[0], d_in[13], S * sizeof(float), hipMemcpyDeviceToDevice, stream);
  hipMemcpyAsync(c1b[0], d_in[14], S * sizeof(float), hipMemcpyDeviceToDevice, stream);
  hipMemcpyAsync(h2b[0], d_in[15], S * sizeof(float), hipMemcpyDeviceToDevice, stream);
  hipMemcpyAsync(c2b[0], d_in[16], S * sizeof(float), hipMemcpyDeviceToDevice, stream);
  hipMemcpyAsync(h3b[0], d_in[17], S * sizeof(float), hipMemcpyDeviceToDevice, stream);
  hipMemcpyAsync(c3b[0], d_in[18], S * sizeof(float), hipMemcpyDeviceToDevice, stream);

  dim3 grid1(4, 8);      // 4 hidden tiles x 8 batch tiles
  dim3 grid2(4, 8, 2);   // z selects LSTM stream 2 vs 3

  for (int t = 0; t < SEQ_LEN; ++t) {
    int rp = t & 1, wp = rp ^ 1;
    lstm_step1<<<grid1, 256, 0, stream>>>(
        x + (size_t)t * BATCH * IN_DIM, Wih1, Whh1, bih1, bhh1,
        h1b[rp], c1b[rp], h1b[wp], c1b[wp]);
    lstm_step23<<<grid2, 256, 0, stream>>>(
        h1b[wp], Wih2, Whh2, bih2, bhh2,
        h2b[rp], c2b[rp], h2b[wp], c2b[wp],
        h3b[rp], c3b[rp], h3b[wp], c3b[wp]);
  }

  // final states are in parity buffer 1 (SEQ_LEN odd)
  fc1_kernel<<<(BATCH * FC1N + 255) / 256, 256, 0, stream>>>(h3b[1], Wfc1, bfc1, hid);
  fc2_kernel<<<(BATCH * OUTN + 255) / 256, 256, 0, stream>>>(hid, Wfc2, bfc2, (float*)d_out);
}
